// CrossMultiHead_51410758533539
// MI455X (gfx1250) — matrix operand order repaired and checked by reading, not compiled, not run on hardware
//
#include <hip/hip_runtime.h>

typedef __attribute__((ext_vector_type(16))) _Float16 v16h;
typedef __attribute__((ext_vector_type(8)))  _Float16 v8h;
typedef __attribute__((ext_vector_type(8)))  float    v8f;
typedef __attribute__((ext_vector_type(4)))  float    v4f;

#define WMMA_F16(A, B, C) \
  __builtin_amdgcn_wmma_f32_16x16x32_f16(false, (A), false, (B), (short)0, (C), false, false)

#if __has_builtin(__builtin_amdgcn_global_load_async_to_lds_b128)
#define HAVE_ASYNC_LDS 1
typedef __attribute__((ext_vector_type(4))) int i4;
typedef __attribute__((address_space(1))) i4* g_i4p;   // global int4*
typedef __attribute__((address_space(3))) i4* l_i4p;   // LDS int4*
#if __has_builtin(__builtin_amdgcn_s_wait_asynccnt)
#define WAIT_ASYNC() __builtin_amdgcn_s_wait_asynccnt(0)
#else
#define WAIT_ASYNC() asm volatile("s_wait_asynccnt 0x0" ::: "memory")
#endif
#endif

static constexpr int Bsz = 2, S = 2048, Dm = 1024, H = 16, DI = 64;

// ---------------- weight convert + transpose: Wt[n*K + k] = (f16) W[k*N + n] ----------------
__global__ void wt_kernel(const float* __restrict__ W, _Float16* __restrict__ Wt, int K, int N) {
  int idx = blockIdx.x * 256 + threadIdx.x;
  if (idx >= K * N) return;
  int n = idx % N, k = idx / N;                  // coalesced read of W
  Wt[(size_t)n * K + k] = (_Float16)W[idx];
}

// ---------------- mask transpose: maskT[b][k][q] = mask[b][q][k] -----------------------------
__global__ void mt_kernel(const unsigned char* __restrict__ mask, unsigned char* __restrict__ maskT) {
  int idx = blockIdx.x * 256 + threadIdx.x;      // over B*S*S, q fastest (coalesced write)
  int q = idx & (S - 1);
  int rest = idx >> 11;
  int k = rest & (S - 1);
  int b = rest >> 11;
  maskT[idx] = mask[((size_t)b * S + q) * S + k];
}

// ---------------- WMMA GEMM: C[M,N] = A[M,K] * Bt[N,K]^T ; one wave = 16x64 output strip ------
// MODE 0: plain store to C[M,N].  MODE 1: KV split -> K half to C ([M,N/2]), V half scattered
// transposed into Vt[B*H*DI, S].
template <typename AT, typename CT, int MODE>
__global__ void gemm_kernel(const AT* __restrict__ A, const _Float16* __restrict__ Bt,
                            CT* __restrict__ C, _Float16* __restrict__ Vt,
                            int M, int N, int K) {
  const int lane = threadIdx.x & 31;
  const int wid  = (blockIdx.x * 256 + threadIdx.x) >> 5;
  const int tiles_n = N >> 6;
  const int wm = wid / tiles_n, wn = wid % tiles_n;
  const int m = lane & 15, kg = lane >> 4;
  const int arow = wm * 16 + m;

  v8f acc[4] = {};
  for (int k = 0; k < K; k += 32) {
    v16h a;
    if constexpr (sizeof(AT) == 4) {             // f32 A: load 2x8 floats, convert in-register
      const float* ap = (const float*)A + (size_t)arow * K + k + 16 * kg;
      v4f x0 = *(const v4f*)ap,        x1 = *(const v4f*)(ap + 4);
      v4f x2 = *(const v4f*)(ap + 8), x3 = *(const v4f*)(ap + 12);
#pragma unroll
      for (int e = 0; e < 4; e++) {
        a[e]      = (_Float16)x0[e];  a[4 + e]  = (_Float16)x1[e];
        a[8 + e]  = (_Float16)x2[e];  a[12 + e] = (_Float16)x3[e];
      }
    } else {                                     // f16 A: two contiguous 16B chunks
      const _Float16* ap = (const _Float16*)A + (size_t)arow * K + k + 16 * kg;
      v8h h0 = *(const v8h*)ap, h1 = *(const v8h*)(ap + 8);
#pragma unroll
      for (int e = 0; e < 8; e++) { a[e] = h0[e]; a[8 + e] = h1[e]; }
    }
#pragma unroll
    for (int nt = 0; nt < 4; nt++) {             // B-frag: contiguous 32B from transposed weight
      const _Float16* bp = Bt + (size_t)(wn * 64 + nt * 16 + m) * K + k + 16 * kg;
      v16h b = *(const v16h*)bp;
      acc[nt] = WMMA_F16(a, b, acc[nt]);
    }
  }
#pragma unroll
  for (int nt = 0; nt < 4; nt++)
#pragma unroll
    for (int r = 0; r < 8; r++) {
      int row = wm * 16 + 8 * kg + r;
      int col = wn * 64 + nt * 16 + m;
      if constexpr (MODE == 0) {
        C[(size_t)row * N + col] = (CT)acc[nt][r];
      } else {
        if (col < (N >> 1)) {                    // K half -> compact [M, N/2]
          C[(size_t)row * (N >> 1) + col] = (CT)acc[nt][r];
        } else {                                 // V half -> transposed [B*H*DI, S]
          int c = col - (N >> 1);
          int h = c >> 6, di = c & (DI - 1);
          int b = row >> 11, s = row & (S - 1);
          Vt[(size_t)((b * H + h) * DI + di) * S + s] = (_Float16)acc[nt][r];
        }
      }
    }
}

// ---------------- flash attention -----------------------------------------------------------
// One block = 8 waves, all on the same (b,h), 128 queries. K/V 32-key tiles are staged into
// LDS once per block (double-buffered, 1 barrier/step; async global->LDS when available).
__global__ void attn_kernel(const _Float16* __restrict__ Q, const _Float16* __restrict__ Kb,
                            const _Float16* __restrict__ Vt, const unsigned char* __restrict__ maskT,
                            _Float16* __restrict__ Z) {
  __shared__ __align__(16) _Float16 KS[2][32 * 72];    // [key][di], padded stride 72
  __shared__ __align__(16) _Float16 VS[2][64 * 40];    // [di][key], padded stride 40
  __shared__ __align__(16) _Float16 PS[8][16 * 40];    // per-wave P tile

  const int tid = threadIdx.x;
  const int lane = tid & 31;
  const int wslot = tid >> 5;
  const int qb = blockIdx.x & 15;                      // 16 q-blocks of 128
  const int h  = (blockIdx.x >> 4) & (H - 1);
  const int b  = blockIdx.x >> 8;
  const int m = lane & 15, kg = lane >> 4;
  const float scale = 0.125f;                          // DI^-0.5
  const int q0 = qb * 128 + wslot * 16;                // this wave's first query row

  // --- staging addresses (whole block cooperates; fully coalesced 16B chunks)
  const int krow = tid >> 3, kc = tid & 7;             // K tile: 32 rows x 8 chunks
  const _Float16* kgp = Kb + (size_t)(b * S + krow) * (H * DI) + h * DI + kc * 8;
  _Float16* ksp = &KS[0][krow * 72 + kc * 8];
  const int vrow = tid >> 2, vc = tid & 3;             // V tile: 64 rows x 4 chunks
  const _Float16* vgp = Vt + (size_t)((b * H + h) * DI + vrow) * S + vc * 8;
  _Float16* vsp = &VS[0][vrow * 40 + vc * 8];

  // --- Q A-fragments for di in [0,32) and [32,64), loaded once
  const _Float16* qp = Q + (size_t)(b * S + q0 + m) * (H * DI) + h * DI;
  v16h aq0, aq1;
  {
    v8h qv0 = *(const v8h*)(qp + 16 * kg);
    v8h qv1 = *(const v8h*)(qp + 8 + 16 * kg);
    v8h qv2 = *(const v8h*)(qp + 32 + 16 * kg);
    v8h qv3 = *(const v8h*)(qp + 40 + 16 * kg);
#pragma unroll
    for (int e = 0; e < 8; e++) { aq0[e] = qv0[e]; aq0[8 + e] = qv1[e];
                                  aq1[e] = qv2[e]; aq1[8 + e] = qv3[e]; }
  }

  float mrow[8], lrow[8];
  v8f zacc[4] = {};
#pragma unroll
  for (int r = 0; r < 8; r++) { mrow[r] = -INFINITY; lrow[r] = 0.0f; }

  _Float16* pl = PS[wslot];
  const unsigned char* mtp = maskT + (size_t)b * S * S + q0 + 8 * kg;

  // --- preload tile 0
#if defined(HAVE_ASYNC_LDS)
  __builtin_amdgcn_global_load_async_to_lds_b128((g_i4p)kgp, (l_i4p)ksp, 0, 0);
  __builtin_amdgcn_global_load_async_to_lds_b128((g_i4p)vgp, (l_i4p)vsp, 0, 0);
  WAIT_ASYNC();
#else
  {
    v8h kv8 = *(const v8h*)kgp;
    v8h vv8 = *(const v8h*)vgp;
    *(v8h*)ksp = kv8;
    *(v8h*)vsp = vv8;
  }
#endif
  __syncthreads();

  constexpr int NSTEP = S / 32;                        // 64 key steps
  for (int step = 0; step < NSTEP; step++) {
    const int cur = step & 1;
    const int key0 = step * 32;
    // ---- stage next tile into the other buffer (drained before end-of-step barrier)
    if (step + 1 < NSTEP) {
#if defined(HAVE_ASYNC_LDS)
      __builtin_amdgcn_global_load_async_to_lds_b128(
          (g_i4p)(kgp + (size_t)(step + 1) * 32 * (H * DI)),
          (l_i4p)(ksp + (cur ^ 1) * 32 * 72), 0, 0);
      __builtin_amdgcn_global_load_async_to_lds_b128(
          (g_i4p)(vgp + (step + 1) * 32),
          (l_i4p)(vsp + (cur ^ 1) * 64 * 40), 0, 0);
#else
      v8h kv8 = *(const v8h*)(kgp + (size_t)(step + 1) * 32 * (H * DI));
      v8h vv8 = *(const v8h*)(vgp + (step + 1) * 32);
      *(v8h*)(ksp + (cur ^ 1) * 32 * 72) = kv8;
      *(v8h*)(vsp + (cur ^ 1) * 64 * 40) = vv8;
#endif
    }
    // ---- mask chunks: lane's 8 q-rows contiguous per key column
    uint2 mb0 = *(const uint2*)(mtp + (size_t)(key0 + m) * S);
    uint2 mb1 = *(const uint2*)(mtp + (size_t)(key0 + 16 + m) * S);
    // ---- scores: two 16x16 tiles; B = K^T read from LDS K tile rows
    v8f sc[2] = {};
#pragma unroll
    for (int t = 0; t < 2; t++) {
      const _Float16* kp = &KS[cur][(t * 16 + m) * 72 + kg * 16];
      v16h bk0 = *(const v16h*)kp;
      v16h bk1 = *(const v16h*)(kp + 32);
      sc[t] = WMMA_F16(aq0, bk0, sc[t]);
      sc[t] = WMMA_F16(aq1, bk1, sc[t]);
    }
    // ---- scale + mask
#pragma unroll
    for (int t = 0; t < 2; t++) {
      unsigned lo = t ? mb1.x : mb0.x;
      unsigned hi = t ? mb1.y : mb0.y;
#pragma unroll
      for (int r = 0; r < 8; r++) {
        float v = sc[t][r] * scale;
        unsigned byte = (r < 4 ? (lo >> (8 * r)) : (hi >> (8 * (r - 4)))) & 0xffu;
        if (byte) v = -INFINITY;
        sc[t][r] = v;
      }
    }
    // ---- online softmax (row stats across 16-lane halves)
#pragma unroll
    for (int r = 0; r < 8; r++) {
      float rm = fmaxf(sc[0][r], sc[1][r]);
#pragma unroll
      for (int off = 1; off <= 8; off <<= 1) rm = fmaxf(rm, __shfl_xor(rm, off, 32));
      float mnew = fmaxf(mrow[r], rm);
      float alpha = __expf(mrow[r] - mnew);
      mrow[r] = mnew;
      float p0 = __expf(sc[0][r] - mnew);
      float p1 = __expf(sc[1][r] - mnew);
      float rs = p0 + p1;
#pragma unroll
      for (int off = 1; off <= 8; off <<= 1) rs += __shfl_xor(rs, off, 32);
      lrow[r] = lrow[r] * alpha + rs;
#pragma unroll
      for (int nt = 0; nt < 4; nt++) zacc[nt][r] *= alpha;
      pl[(8 * kg + r) * 40 + m]      = (_Float16)p0;   // P[q][key] -> LDS (C-frag layout)
      pl[(8 * kg + r) * 40 + 16 + m] = (_Float16)p1;
    }
    // ---- re-read P as A-fragment (same-wave LDS is in-order)
    v16h ap;
    {
      v8h pa0 = *(const v8h*)(pl + m * 40 + 16 * kg);
      v8h pa1 = *(const v8h*)(pl + m * 40 + 8 + 16 * kg);
#pragma unroll
      for (int e = 0; e < 8; e++) { ap[e] = pa0[e]; ap[8 + e] = pa1[e]; }
    }
    // ---- P (16x32) @ V (32x64): B-frags from LDS V tile rows
#pragma unroll
    for (int nt = 0; nt < 4; nt++) {
      const _Float16* vp = &VS[cur][(nt * 16 + m) * 40 + kg * 16];
      v16h bv = *(const v16h*)vp;
      zacc[nt] = WMMA_F16(ap, bv, zacc[nt]);
    }
#if defined(HAVE_ASYNC_LDS)
    WAIT_ASYNC();                                      // next tile fully in LDS
#endif
    __syncthreads();                                   // one barrier per key step
  }
  // ---- normalize and store Z (f16) in [B*S, H*DI] layout
#pragma unroll
  for (int r = 0; r < 8; r++) {
    float inv = 1.0f / lrow[r];
    int row = b * S + q0 + 8 * kg + r;
#pragma unroll
    for (int nt = 0; nt < 4; nt++)
      Z[(size_t)row * (H * DI) + h * DI + nt * 16 + m] = (_Float16)(zacc[nt][r] * inv);
  }
}

extern "C" void kernel_launch(void* const* d_in, const int* in_sizes, int n_in,
                              void* d_out, int out_size, void* d_ws, size_t ws_size,
                              hipStream_t stream) {
  const float* x            = (const float*)d_in[0];
  const float* kv           = (const float*)d_in[1];
  const unsigned char* mask = (const unsigned char*)d_in[2];  // all-false bools
  const float* Wq           = (const float*)d_in[3];
  const float* Wkv          = (const float*)d_in[4];
  const float* Wz           = (const float*)d_in[5];
  float* out = (float*)d_out;

  char* ws = (char*)d_ws;                       // 48 MiB total
  _Float16* Wqt      = (_Float16*)(ws);                        //  2 MiB: [1024,1024]
  _Float16* Wkvt     = (_Float16*)(ws + (size_t)2  * 1048576); //  4 MiB: [2048,1024]
  _Float16* Wzt      = (_Float16*)(ws + (size_t)6  * 1048576); //  2 MiB: [1024,1024]
  _Float16* Qb       = (_Float16*)(ws + (size_t)8  * 1048576); //  8 MiB: [4096,1024]
  _Float16* Kbuf     = (_Float16*)(ws + (size_t)16 * 1048576); //  8 MiB: [4096,1024]
  _Float16* Vtb      = (_Float16*)(ws + (size_t)24 * 1048576); //  8 MiB: [B*H*DI, S]
  _Float16* Zb       = (_Float16*)(ws + (size_t)32 * 1048576); //  8 MiB: [4096,1024]
  unsigned char* mTb = (unsigned char*)(ws + (size_t)40 * 1048576); // 8 MiB: [B,S,S]

  // 1) weights -> f16, transposed to [N,K]; mask -> transposed bytes
  wt_kernel<<<(1024 * 1024) / 256, 256, 0, stream>>>(Wq,  Wqt,  1024, 1024);
  wt_kernel<<<(1024 * 2048) / 256, 256, 0, stream>>>(Wkv, Wkvt, 1024, 2048);
  wt_kernel<<<(1024 * 1024) / 256, 256, 0, stream>>>(Wz,  Wzt,  1024, 1024);
  mt_kernel<<<(Bsz * S * S) / 256, 256, 0, stream>>>(mask, mTb);
  // 2) projections (f32 A converted in-kernel); KV GEMM scatters V transposed
  gemm_kernel<float, _Float16, 0><<<(256 * 16) / 8, 256, 0, stream>>>(x,  Wqt,  Qb,   nullptr, 4096, 1024, 1024);
  gemm_kernel<float, _Float16, 1><<<(256 * 32) / 8, 256, 0, stream>>>(kv, Wkvt, Kbuf, Vtb,     4096, 2048, 1024);
  // 3) attention: 512 blocks of 8 waves sharing K/V tiles via LDS
  attn_kernel<<<Bsz * H * (S / 128), 256, 0, stream>>>(Qb, Kbuf, Vtb, mTb, Zb);
  // 4) output projection -> f32
  gemm_kernel<_Float16, float, 0><<<(256 * 16) / 8, 256, 0, stream>>>(Zb, Wzt, out, nullptr, 4096, 1024, 1024);
}
